// SigLIPLoss_60275571032311
// MI455X (gfx1250) — matrix operand order repaired and checked by reading, not compiled, not run on hardware
//
#include <hip/hip_runtime.h>
#include <hip/hip_bf16.h>
#include <stdint.h>

// ---------------------------------------------------------------------------
// SigLIP loss, fused: sim = z_a @ z_b^T  ->  -mean(logsigmoid(targets*sim+b))
// Split-bf16 (hi/lo) WMMA GEMM on v_wmma_f32_16x16x32_bf16.
// B pre-transposed once; panels staged with global_load_async_to_lds_b128.
// ---------------------------------------------------------------------------

typedef __attribute__((ext_vector_type(16))) __bf16 v16bf;
typedef __attribute__((ext_vector_type(8)))  float  v8f;

constexpr int N      = 8192;
constexpr int D      = 256;
constexpr int TILE   = 128;   // output tile edge per workgroup
constexpr int KC     = 32;    // K chunk per WMMA pass
constexpr int APITCH = 40;    // LDS A row pitch (bf16 elems): 80 B, 16B-aligned
constexpr int BPITCH = 136;   // LDS B^T row pitch (bf16 elems): 272 B, 16B-aligned
constexpr int GTILES = (N / TILE) * (N / TILE);  // 4096 blocks

struct U32x8 { uint4 lo, hi; };

static __device__ inline v16bf make_frag(uint4 a, uint4 b) {
  U32x8 t{a, b};
  return __builtin_bit_cast(v16bf, t);   // 32 bytes -> 16 x bf16
}

static __device__ inline unsigned short f32_to_bf16_rne(float f) {
  unsigned u = __builtin_bit_cast(unsigned, f);
  unsigned r = u + 0x7FFFu + ((u >> 16) & 1u);   // round-to-nearest-even
  return (unsigned short)(r >> 16);
}
static __device__ inline float bf16_to_f32(unsigned short h) {
  unsigned u = ((unsigned)h) << 16;
  return __builtin_bit_cast(float, u);
}

// Async copy: global -> LDS, 16 bytes per lane, tracked by ASYNCcnt.
// GVS mode: 64-bit SGPR base + per-lane 32-bit byte offset; LDS dest VGPR.
static __device__ inline void async_ld_b128(unsigned lds_byte, unsigned gbyte,
                                            const void* base) {
  asm volatile("global_load_async_to_lds_b128 %0, %1, %2"
               :: "v"(lds_byte), "v"(gbyte), "s"((unsigned long long)(uintptr_t)base)
               : "memory");
}
static __device__ inline void wait_async0() {
  asm volatile("s_wait_asynccnt 0x0" ::: "memory");
}

// ---- Kernel 1a: split z_a (f32) -> hi/lo bf16, row-major [N][D] ------------
__global__ __launch_bounds__(256) void siglip_cvt_a(
    const float* __restrict__ za,
    unsigned short* __restrict__ aH, unsigned short* __restrict__ aL) {
  const int i = blockIdx.x * 256 + threadIdx.x;
  float x = za[i];
  unsigned short h = f32_to_bf16_rne(x);
  unsigned short l = f32_to_bf16_rne(x - bf16_to_f32(h));
  aH[i] = h;
  aL[i] = l;
}

// ---- Kernel 1b: split + TRANSPOSE z_b -> hi/lo bf16 in [D][N] layout -------
// 64x64 tiles through LDS so both global read and write are coalesced.
__global__ __launch_bounds__(256) void siglip_cvt_b_t(
    const float* __restrict__ zb,
    unsigned short* __restrict__ bHt, unsigned short* __restrict__ bLt) {
  __shared__ unsigned short tH[64][72];
  __shared__ unsigned short tL[64][72];
  const int tid = threadIdx.x;
  const int n0  = (blockIdx.x >> 2) * 64;   // 128 n-tiles
  const int k0  = (blockIdx.x & 3) * 64;    // 4 k-tiles
  {
    const int r   = tid >> 2;               // local n row
    const int c16 = (tid & 3) * 16;         // local k base
    const float* src = zb + (size_t)(n0 + r) * D + k0 + c16;
#pragma unroll
    for (int e = 0; e < 16; ++e) {
      float x = src[e];
      unsigned short h = f32_to_bf16_rne(x);
      unsigned short l = f32_to_bf16_rne(x - bf16_to_f32(h));
      tH[r][c16 + e] = h;
      tL[r][c16 + e] = l;
    }
  }
  __syncthreads();
  {
    const int kk = tid >> 2;                // local k row of output
    const int nb = (tid & 3) * 16;          // local n base
    alignas(16) unsigned short oh[16], ol[16];
#pragma unroll
    for (int e = 0; e < 16; ++e) {
      oh[e] = tH[nb + e][kk];
      ol[e] = tL[nb + e][kk];
    }
    unsigned short* dh = bHt + (size_t)(k0 + kk) * N + n0 + nb;
    unsigned short* dl = bLt + (size_t)(k0 + kk) * N + n0 + nb;
    *(uint4*)dh       = *(const uint4*)oh;
    *(uint4*)(dh + 8) = *(const uint4*)(oh + 8);
    *(uint4*)dl       = *(const uint4*)ol;
    *(uint4*)(dl + 8) = *(const uint4*)(ol + 8);
  }
}

// ---- Kernel 2: fused tile GEMM (split-bf16 WMMA) + logsigmoid reduction ----
__global__ __launch_bounds__(256) void siglip_gemm_loss(
    const unsigned short* __restrict__ aH,  const unsigned short* __restrict__ aL,
    const unsigned short* __restrict__ bHt, const unsigned short* __restrict__ bLt,
    const float* __restrict__ bias_p, float* __restrict__ partials) {
  __shared__ unsigned short sAh[TILE][APITCH];   // A panel (hi), row-major
  __shared__ unsigned short sAl[TILE][APITCH];   // A panel (lo)
  __shared__ unsigned short sBh[KC][BPITCH];     // B^T panel (hi): [k][n]
  __shared__ unsigned short sBl[KC][BPITCH];     // B^T panel (lo)
  __shared__ float rbuf[256];

  const int tid     = threadIdx.x;
  const int lane    = tid & 31;
  const int w       = tid >> 5;                  // wave w owns rows 16w..16w+15
  const int rowBase = (blockIdx.x >> 6) * TILE;  // 64x64 tile grid
  const int colBase = (blockIdx.x & 63) * TILE;

  const unsigned ldsAh = (unsigned)(uintptr_t)&sAh[0][0];
  const unsigned ldsAl = (unsigned)(uintptr_t)&sAl[0][0];
  const unsigned ldsBh = (unsigned)(uintptr_t)&sBh[0][0];
  const unsigned ldsBl = (unsigned)(uintptr_t)&sBl[0][0];

  v8f acc[8];
#pragma unroll
  for (int j = 0; j < 8; ++j) acc[j] = v8f{0.f, 0.f, 0.f, 0.f, 0.f, 0.f, 0.f, 0.f};

  for (int k0 = 0; k0 < D; k0 += KC) {
    // ---- async global -> LDS panel stage (no VGPR round-trip) ----
#pragma unroll
    for (int c = 0; c < 2; ++c) {
      const int chunk = tid * 2 + c;              // 0..511 x 16B = 8 KB panel
      {                                           // A: 4 chunks per 64 B row
        const int row  = chunk >> 2;
        const int colb = (chunk & 3) * 16;        // byte offset in row
        const unsigned gb =
            (unsigned)(((rowBase + row) * D + k0) * 2) + (unsigned)colb;
        const unsigned lb = (unsigned)(row * (APITCH * 2) + colb);
        async_ld_b128(ldsAh + lb, gb, aH);
        async_ld_b128(ldsAl + lb, gb, aL);
      }
      {                                           // B^T: 16 chunks per 256 B row
        const int row  = chunk >> 4;              // k row 0..31
        const int colb = (chunk & 15) * 16;
        const unsigned gb =
            (unsigned)(((k0 + row) * N + colBase) * 2) + (unsigned)colb;
        const unsigned lb = (unsigned)(row * (BPITCH * 2) + colb);
        async_ld_b128(ldsBh + lb, gb, bHt);
        async_ld_b128(ldsBl + lb, gb, bLt);
      }
    }
    wait_async0();
    __syncthreads();

    // ---- A fragments: lane<16 -> M=lane, K {0..7,16..23}; lane>=16 -> K {8..15,24..31}
    const int m    = lane & 15;
    const int koff = (lane < 16) ? 0 : 8;
    const unsigned short* pa  = &sAh[16 * w + m][koff];
    const unsigned short* pal = &sAl[16 * w + m][koff];
    v16bf fAh = make_frag(*(const uint4*)pa,  *(const uint4*)(pa + 16));
    v16bf fAl = make_frag(*(const uint4*)pal, *(const uint4*)(pal + 16));

#pragma unroll
    for (int j = 0; j < 8; ++j) {
      // B fragment: lane -> N = lane%16, elems K {koff..koff+7, 16+koff..16+koff+7}
      const int nn = 16 * j + m;
      v16bf fBh, fBl;
#pragma unroll
      for (int s = 0; s < 16; ++s) {
        const int kk = (s < 8) ? (koff + s) : (16 + koff + (s - 8));
        fBh[s] = __builtin_bit_cast(__bf16, sBh[kk][nn]);
        fBl[s] = __builtin_bit_cast(__bf16, sBl[kk][nn]);
      }
      // split-bf16: hi*hi + hi*lo + lo*hi  (~f32 accuracy on bf16 pipes)
      acc[j] = __builtin_amdgcn_wmma_f32_16x16x32_bf16(false, fAh, false, fBh,
                                                       (short)0, acc[j], false, false);
      acc[j] = __builtin_amdgcn_wmma_f32_16x16x32_bf16(false, fAh, false, fBl,
                                                       (short)0, acc[j], false, false);
      acc[j] = __builtin_amdgcn_wmma_f32_16x16x32_bf16(false, fAl, false, fBh,
                                                       (short)0, acc[j], false, false);
    }
    __syncthreads();   // all waves done reading before next panel overwrite
  }

  // ---- fused epilogue: logits = targets*sim + bias; sum logsigmoid ----
  const float bias = *bias_p;
  float lsum = 0.f;
  const int mg = rowBase + 16 * w + ((lane >= 16) ? 8 : 0);  // C/D: VGPR r -> M=r(+8)
  const int ng = colBase + (lane & 15);                       // lane -> N
#pragma unroll
  for (int j = 0; j < 8; ++j) {
    const int n = ng + 16 * j;
#pragma unroll
    for (int r = 0; r < 8; ++r) {
      const float s  = acc[j][r];
      const int   mm = mg + r;
      const float t  = ((mm == n) ? s : -s) + bias;            // +1 diag, -1 off
      lsum += fminf(t, 0.f) - __logf(1.f + __expf(-fabsf(t))); // stable logsigmoid
    }
  }

  // deterministic block reduction (fixed tree order -> graph-replay stable)
  rbuf[tid] = lsum;
  __syncthreads();
#pragma unroll
  for (int off = 128; off > 0; off >>= 1) {
    if (tid < off) rbuf[tid] += rbuf[tid + off];
    __syncthreads();
  }
  if (tid == 0) partials[blockIdx.x] = rbuf[0];
}

// ---- Kernel 3: deterministic final reduction over 4096 partials ------------
__global__ __launch_bounds__(256) void siglip_finalize(
    const float* __restrict__ partials, float* __restrict__ out) {
  __shared__ float rbuf[256];
  const int t = threadIdx.x;
  float s = 0.f;
#pragma unroll
  for (int i = 0; i < GTILES / 256; ++i) s += partials[t + 256 * i];
  rbuf[t] = s;
  __syncthreads();
#pragma unroll
  for (int off = 128; off > 0; off >>= 1) {
    if (t < off) rbuf[t] += rbuf[t + off];
    __syncthreads();
  }
  if (t == 0) out[0] = -rbuf[0] / (float)((long long)N * (long long)N);
}

// ---------------------------------------------------------------------------
extern "C" void kernel_launch(void* const* d_in, const int* in_sizes, int n_in,
                              void* d_out, int out_size, void* d_ws, size_t ws_size,
                              hipStream_t stream) {
  (void)in_sizes; (void)n_in; (void)out_size; (void)ws_size;
  const float* za   = (const float*)d_in[0];
  const float* zb   = (const float*)d_in[1];
  const float* bias = (const float*)d_in[2];
  float* out = (float*)d_out;

  char* ws = (char*)d_ws;
  float* partials = (float*)ws;                                   // 4096 f32 = 16 KB
  const size_t ND = (size_t)N * (size_t)D;
  unsigned short* aH  = (unsigned short*)(ws + 16384);            // [N][D] hi
  unsigned short* aL  = aH + ND;                                  // [N][D] lo
  unsigned short* bHt = aL + ND;                                  // [D][N] hi (transposed)
  unsigned short* bLt = bHt + ND;                                 // [D][N] lo (transposed)

  siglip_cvt_a<<<(N * D) / 256, 256, 0, stream>>>(za, aH, aL);
  siglip_cvt_b_t<<<(N / 64) * (D / 64), 256, 0, stream>>>(zb, bHt, bLt);
  siglip_gemm_loss<<<GTILES, 256, 0, stream>>>(aH, aL, bHt, bLt, bias, partials);
  siglip_finalize<<<1, 256, 0, stream>>>(partials, out);
}
